// Model_71648644432342
// MI455X (gfx1250) — compile-verified, matrix operand order checked by reading, not run on hardware
//
#include <hip/hip_runtime.h>
#include <hip/hip_bf16.h>
#include <math.h>

#define BB      64
#define DD      64
#define SEQL    512
#define PATCH   32
#define DM      256
#define DSTATE  16
#define DCONV   4
#define DINNER  512
#define DTRANK  16
#define IN_SEG  16
#define OUT_SEG 8

typedef __attribute__((ext_vector_type(16))) __bf16 v16bf;
typedef __attribute__((ext_vector_type(8)))  __bf16 v8bf;
typedef __attribute__((ext_vector_type(4)))  __bf16 v4bf;
typedef __attribute__((ext_vector_type(8)))  float  v8f;

static __device__ __forceinline__ float epi(float v, const float* bias, int gn, int fuse) {
  if (fuse & 1) v += bias[gn];
  if (fuse & 2) v = 0.5f * v * (1.f + erff(v * 0.70710678118f));          // gelu
  if (fuse & 4) v = (v > 20.f) ? v : log1pf(__expf(v));                   // softplus
  return v;
}
static __device__ __forceinline__ float silu(float v) { return v / (1.f + __expf(-v)); }

// ---------------------------------------------------------------------------
// FAST GEMM: M%64==0, N%64==0, K%32==0. Bt = bf16 weights pre-transposed
// [N][K]. 128 threads = 4 waves (2x2), wave tile 32x32 -> 4 WMMA / K-step.
// Double-buffered LDS; panels loaded to registers first (batched b128 loads),
// next panel's loads overlap current panel's WMMA. Fragments = 2x ds_load_b128.
// ---------------------------------------------------------------------------
__global__ __launch_bounds__(128)
void gemm_wmma_fast(const float* __restrict__ A, int lda,
                    const __bf16* __restrict__ Bt,
                    const float* __restrict__ bias,
                    float* __restrict__ C, int ldc,
                    int M, int N, int K, int fuse) {
  __shared__ __bf16 As[2][64][32];
  __shared__ __bf16 Bs[2][64][32];
  const int tid  = threadIdx.x & 127;          // provable range -> no guards
  const int wave = tid >> 5, lane = tid & 31;
  const int wm = (wave >> 1) * 32;
  const int wn = (wave & 1) * 32;
  const int m0 = blockIdx.x * 64;
  const int n0 = blockIdx.y * 64;
  const int hlf = lane >> 4, mr = lane & 15;

  int rA[4], cA[4], rB[2], cB[2];
#pragma unroll
  for (int u = 0; u < 4; ++u) { int i = tid + u * 128; rA[u] = i >> 3; cA[u] = (i << 2) & 31; }
#pragma unroll
  for (int u = 0; u < 2; ++u) { int i = tid + u * 128; rB[u] = i >> 2; cB[u] = (i << 3) & 31; }

  float4 ga[4];
  v8bf   gb[2];
  auto load_panel = [&](int k0) {
#pragma unroll
    for (int u = 0; u < 4; ++u)
      ga[u] = *(const float4*)&A[(size_t)(m0 + rA[u]) * lda + k0 + cA[u]];
#pragma unroll
    for (int u = 0; u < 2; ++u)
      gb[u] = *(const v8bf*)&Bt[(size_t)(n0 + rB[u]) * K + k0 + cB[u]];
  };
  load_panel(0);

  v8f acc[2][2] = {};
  int buf = 0;
  for (int k0 = 0; k0 < K; k0 += 32) {
#pragma unroll
    for (int u = 0; u < 4; ++u) {
      v4bf pk;
      pk[0] = (__bf16)ga[u].x; pk[1] = (__bf16)ga[u].y;
      pk[2] = (__bf16)ga[u].z; pk[3] = (__bf16)ga[u].w;
      *(v4bf*)&As[buf][rA[u]][cA[u]] = pk;
    }
#pragma unroll
    for (int u = 0; u < 2; ++u)
      *(v8bf*)&Bs[buf][rB[u]][cB[u]] = gb[u];
    __syncthreads();
    if (k0 + 32 < K) load_panel(k0 + 32);      // overlap with WMMA below
    v16bf fa[2], fb[2];
#pragma unroll
    for (int a = 0; a < 2; ++a) {
      v8bf lo = *(const v8bf*)&As[buf][wm + a * 16 + mr][hlf * 8];
      v8bf hi = *(const v8bf*)&As[buf][wm + a * 16 + mr][16 + hlf * 8];
      fa[a] = __builtin_shufflevector(lo, hi, 0,1,2,3,4,5,6,7,8,9,10,11,12,13,14,15);
      v8bf bl = *(const v8bf*)&Bs[buf][wn + a * 16 + mr][hlf * 8];
      v8bf bh = *(const v8bf*)&Bs[buf][wn + a * 16 + mr][16 + hlf * 8];
      fb[a] = __builtin_shufflevector(bl, bh, 0,1,2,3,4,5,6,7,8,9,10,11,12,13,14,15);
    }
#pragma unroll
    for (int a = 0; a < 2; ++a)
#pragma unroll
      for (int b = 0; b < 2; ++b)
        acc[a][b] = __builtin_amdgcn_wmma_f32_16x16x32_bf16(
            false, fa[a], false, fb[b], (short)0, acc[a][b], false, false);
    buf ^= 1;
  }
  const int col = lane & 15, rh = lane >> 4;
#pragma unroll
  for (int a = 0; a < 2; ++a)
#pragma unroll
    for (int b = 0; b < 2; ++b)
#pragma unroll
      for (int r = 0; r < 8; ++r) {
        int gm = m0 + wm + a * 16 + r + rh * 8;
        int gn = n0 + wn + b * 16 + col;
        C[(size_t)gm * ldc + gn] = epi(acc[a][b][r], bias, gn, fuse);
      }
}

// weight -> bf16 transposed [N][K]
__global__ __launch_bounds__(256)
void wt_bf16_k(const float* __restrict__ W, __bf16* __restrict__ Bt, int K, int N) {
  size_t i = (size_t)blockIdx.x * 256 + threadIdx.x;
  if (i >= (size_t)K * N) return;
  int k = (int)(i % K);
  int n = (int)(i / K);
  Bt[i] = (__bf16)W[(size_t)k * N + n];
}

// ---------------------------------------------------------------------------
// GENERAL GEMM (ragged shapes: N=48, K=16, N=32). Registers-first staging with
// clamped unconditional loads + select-zero (no exec branching, batched loads).
// ---------------------------------------------------------------------------
__global__ __launch_bounds__(128)
void gemm_bf16_wmma(const float* __restrict__ A, int lda,
                    const float* __restrict__ Bw, int ldb,
                    const float* __restrict__ bias,
                    float* __restrict__ C, int ldc,
                    int M, int N, int K, int fuse) {
  __shared__ __bf16 As[32][34];
  __shared__ __bf16 Bs[32][34];
  const int tid  = threadIdx.x & 127;
  const int wave = tid >> 5, lane = tid & 31;
  const int wm = (wave >> 1) * 16;
  const int wn = (wave & 1) * 16;
  const int m0 = blockIdx.x * 32;
  const int n0 = blockIdx.y * 32;
  v8f acc = {};
  for (int k0 = 0; k0 < K; k0 += 32) {
    float ra[8], rb[8];
#pragma unroll
    for (int u = 0; u < 8; ++u) {             // batched loads, clamped addresses
      int i = tid + u * 128;
      int r = i >> 5, cc = i & 31;
      int gm = m0 + r, gk = k0 + cc;
      ra[u] = A[(size_t)(gm < M ? gm : M - 1) * lda + (gk < K ? gk : K - 1)];
      int bk = k0 + r, bn = n0 + cc;
      rb[u] = Bw[(size_t)(bk < K ? bk : K - 1) * ldb + (bn < N ? bn : N - 1)];
    }
#pragma unroll
    for (int u = 0; u < 8; ++u) {
      int i = tid + u * 128;
      int r = i >> 5, cc = i & 31;
      As[r][cc] = (__bf16)(((m0 + r) < M && (k0 + cc) < K) ? ra[u] : 0.f);
      Bs[r][cc] = (__bf16)(((k0 + r) < K && (n0 + cc) < N) ? rb[u] : 0.f);
    }
    __syncthreads();
    const int hlf = lane >> 4, mr = lane & 15;
    v16bf fa, fb;
#pragma unroll
    for (int vg = 0; vg < 8; ++vg) {
      int kb = (vg < 4 ? vg * 2 : 16 + (vg - 4) * 2) + hlf * 8;
      fa[2 * vg]     = As[wm + mr][kb];
      fa[2 * vg + 1] = As[wm + mr][kb + 1];
      fb[2 * vg]     = Bs[kb][wn + mr];
      fb[2 * vg + 1] = Bs[kb + 1][wn + mr];
    }
    acc = __builtin_amdgcn_wmma_f32_16x16x32_bf16(false, fa, false, fb,
                                                  (short)0, acc, false, false);
    __syncthreads();
  }
  const int col = lane & 15, rh = lane >> 4;
#pragma unroll
  for (int r = 0; r < 8; ++r) {
    int gm = m0 + wm + r + rh * 8;
    int gn = n0 + wn + col;
    if (gm < M && gn < N) C[(size_t)gm * ldc + gn] = epi(acc[r], bias, gn, fuse);
  }
}

// ---------------------------------------------------------------------------
// Depthwise causal conv (width 4) + SiLU; 4 channels per thread (float4 path).
// Also emits silu(z) gate.
// ---------------------------------------------------------------------------
__global__ __launch_bounds__(256)
void conv_silu_k(const float* __restrict__ xz, const float* __restrict__ cw,
                 const float* __restrict__ cb, float* __restrict__ xc,
                 float* __restrict__ zs, int Nb, int L) {
  size_t i = (size_t)blockIdx.x * 256 + threadIdx.x;
  size_t total = (size_t)Nb * L * (DINNER / 4);
  if (i >= total) return;
  int c = (int)(i % (DINNER / 4)) * 4;
  size_t t = i / (DINNER / 4);
  int l = (int)(t % L);
  size_t n = t / L;
  float4 w0 = *(const float4*)&cw[(c + 0) * DCONV];
  float4 w1 = *(const float4*)&cw[(c + 1) * DCONV];
  float4 w2 = *(const float4*)&cw[(c + 2) * DCONV];
  float4 w3 = *(const float4*)&cw[(c + 3) * DCONV];
  const float wt0[4] = {w0.x, w0.y, w0.z, w0.w};
  const float wt1[4] = {w1.x, w1.y, w1.z, w1.w};
  const float wt2[4] = {w2.x, w2.y, w2.z, w2.w};
  const float wt3[4] = {w3.x, w3.y, w3.z, w3.w};
  float4 a = *(const float4*)&cb[c];
#pragma unroll
  for (int j = 0; j < DCONV; ++j) {
    int ll = l - (DCONV - 1) + j;
    if (ll >= 0) {
      float4 xv = *(const float4*)&xz[((size_t)n * L + ll) * (2 * DINNER) + c];
      a.x += wt0[j] * xv.x; a.y += wt1[j] * xv.y;
      a.z += wt2[j] * xv.z; a.w += wt3[j] * xv.w;
    }
  }
  float4 o; o.x = silu(a.x); o.y = silu(a.y); o.z = silu(a.z); o.w = silu(a.w);
  *(float4*)&xc[t * DINNER + c] = o;
  float4 zv = *(const float4*)&xz[t * (2 * DINNER) + DINNER + c];
  float4 zo; zo.x = silu(zv.x); zo.y = silu(zv.y); zo.z = silu(zv.z); zo.w = silu(zv.w);
  *(float4*)&zs[t * DINNER + c] = zo;
}

// ---------------------------------------------------------------------------
// Selective scan: one thread per (n, channel); 16-state recurrence in regs.
// ---------------------------------------------------------------------------
__global__ __launch_bounds__(256)
void mamba_scan_k(const float* __restrict__ xc, const float* __restrict__ dt,
                  const float* __restrict__ proj, const float* __restrict__ A_log,
                  const float* __restrict__ Dp, float* __restrict__ y,
                  int Nb, int L) {
  size_t idx = (size_t)blockIdx.x * 256 + threadIdx.x;
  if (idx >= (size_t)Nb * DINNER) return;
  int c = (int)(idx % DINNER);
  size_t n = idx / DINNER;
  float Ar[DSTATE], h[DSTATE];
#pragma unroll
  for (int s = 0; s < DSTATE; ++s) { Ar[s] = -__expf(A_log[c * DSTATE + s]); h[s] = 0.f; }
  const float Dc = Dp[c];
  for (int l = 0; l < L; ++l) {
    size_t t = n * L + l;
    float xt  = xc[t * DINNER + c];
    float dtt = dt[t * DINNER + c];
    float dx  = dtt * xt;
    float accv = 0.f;
    const float* pr = proj + t * 48;
#pragma unroll
    for (int s = 0; s < DSTATE; ++s) {
      h[s] = h[s] * __expf(dtt * Ar[s]) + dx * pr[DTRANK + s];
      accv += h[s] * pr[DTRANK + DSTATE + s];
    }
    y[t * DINNER + c] = accv + xt * Dc;
  }
}

// ---------------------------------------------------------------------------
// LayerNorm: one 256-thread block per row of width W (256 or 512).
// ---------------------------------------------------------------------------
__global__ __launch_bounds__(256)
void layernorm_k(const float* __restrict__ in, const float* __restrict__ g,
                 const float* __restrict__ b, float* __restrict__ out, int W) {
  __shared__ float sh0[8], sh1[8];
  const float* x = in + (size_t)blockIdx.x * W;
  float s = 0.f, s2 = 0.f;
  for (int i = threadIdx.x; i < W; i += 256) { float v = x[i]; s += v; s2 += v * v; }
  for (int off = 16; off; off >>= 1) {
    s  += __shfl_down(s, off, 32);
    s2 += __shfl_down(s2, off, 32);
  }
  int wv = threadIdx.x >> 5;
  if ((threadIdx.x & 31) == 0) { sh0[wv] = s; sh1[wv] = s2; }
  __syncthreads();
  if (threadIdx.x == 0) {
    float S = 0.f, S2 = 0.f;
    for (int i = 0; i < 8; ++i) { S += sh0[i]; S2 += sh1[i]; }
    sh0[0] = S; sh1[0] = S2;
  }
  __syncthreads();
  float mu  = sh0[0] / W;
  float var = sh1[0] / W - mu * mu;
  float inv = rsqrtf(var + 1e-5f);
  float* o = out + (size_t)blockIdx.x * W;
  for (int i = threadIdx.x; i < W; i += 256) o[i] = (x[i] - mu) * inv * g[i] + b[i];
}

// (B_,X,Y,DM) -> (B_,Y,X,DM), float4 along DM
__global__ __launch_bounds__(256)
void permute0213(const float* __restrict__ in, float* __restrict__ out,
                 int B_, int X, int Y) {
  size_t i = (size_t)blockIdx.x * 256 + threadIdx.x;
  size_t total = (size_t)B_ * X * Y * (DM / 4);
  if (i >= total) return;
  int dm = (int)(i % (DM / 4)) * 4;
  size_t r = i / (DM / 4);
  int y = (int)(r % Y);
  size_t r2 = r / Y;
  int x = (int)(r2 % X);
  int b = (int)(r2 / X);
  float4 v = *(const float4*)&in[(((size_t)b * X + x) * Y + y) * DM + dm];
  *(float4*)&out[(((size_t)b * Y + y) * X + x) * DM + dm] = v;
}

// x (B*D, 2*S2, DM) -> xm (B*D, S2, 2*DM), float4
__global__ __launch_bounds__(256)
void seg_gather(const float* __restrict__ x, float* __restrict__ xm, int S2) {
  size_t i = (size_t)blockIdx.x * 256 + threadIdx.x;
  size_t total = (size_t)BB * DD * S2 * (2 * DM / 4);
  if (i >= total) return;
  int cc = (int)(i % (2 * DM / 4)) * 4;
  size_t r = i / (2 * DM / 4);
  int s2 = (int)(r % S2);
  size_t bd = r / S2;
  int part = cc / DM, dmc = cc % DM;
  float4 v = *(const float4*)&x[((size_t)bd * (2 * S2) + (2 * s2 + part)) * DM + dmc];
  *(float4*)&xm[(r * 2 * DM) + cc] = v;
}

// x_enc (B,SEQL,D) -> xp (B*D, IN_SEG, PATCH)  (strided gather, scalar)
__global__ __launch_bounds__(256)
void gather_patch(const float* __restrict__ xe, float* __restrict__ xp) {
  size_t i = (size_t)blockIdx.x * 256 + threadIdx.x;
  size_t total = (size_t)BB * DD * IN_SEG * PATCH;
  if (i >= total) return;
  int p = (int)(i % PATCH);
  size_t r = i / PATCH;
  int seg = (int)(r % IN_SEG);
  size_t r2 = r / IN_SEG;
  int d = (int)(r2 % DD);
  int b = (int)(r2 / DD);
  xp[i] = xe[((size_t)b * SEQL + seg * PATCH + p) * DD + d];
}

// x (B,D,OUT_SEG,DM) += linear-interp of enc (B,D,Se,DM), float4
__global__ __launch_bounds__(256)
void add_interp(float* __restrict__ x, const float* __restrict__ e, int Se) {
  size_t i = (size_t)blockIdx.x * 256 + threadIdx.x;
  size_t total = (size_t)BB * DD * OUT_SEG * (DM / 4);
  if (i >= total) return;
  int dm = (int)(i % (DM / 4)) * 4;
  size_t r = i / (DM / 4);
  int s = (int)(r % OUT_SEG);
  size_t bd = r / OUT_SEG;
  float pos = (s + 0.5f) * (Se / (float)OUT_SEG) - 0.5f;
  pos = fminf(fmaxf(pos, 0.f), (float)(Se - 1));
  int i0 = (int)pos;
  int i1 = (i0 + 1 < Se) ? i0 + 1 : Se - 1;
  float w = pos - (float)i0;
  const float* eb = e + (size_t)bd * Se * DM + dm;
  float4 e0 = *(const float4*)&eb[(size_t)i0 * DM];
  float4 e1 = *(const float4*)&eb[(size_t)i1 * DM];
  float4 v = *(float4*)&x[r * DM + dm];
  v.x += e0.x * (1.f - w) + e1.x * w;
  v.y += e0.y * (1.f - w) + e1.y * w;
  v.z += e0.z * (1.f - w) + e1.z * w;
  v.w += e0.w * (1.f - w) + e1.w * w;
  *(float4*)&x[r * DM + dm] = v;
}

// dst[n] (row-broadcast of rows[mod][W]), float4; addf: add vs copy
__global__ __launch_bounds__(256)
void rowbcast(float* __restrict__ dst, const float* __restrict__ rows,
              int mod, int W, size_t n4, int addf) {
  size_t i = (size_t)blockIdx.x * 256 + threadIdx.x;
  if (i >= n4) return;
  int col = (int)(i % (W / 4)) * 4;
  int rr = (int)((i / (W / 4)) % mod);
  float4 v = *(const float4*)&rows[(size_t)rr * W + col];
  if (addf) {
    float4 d = *(float4*)&dst[i * 4];
    v.x += d.x; v.y += d.y; v.z += d.z; v.w += d.w;
  }
  *(float4*)&dst[i * 4] = v;
}

__global__ __launch_bounds__(256)
void ew_add(float* __restrict__ a, const float* __restrict__ b, size_t n4) {
  size_t i = (size_t)blockIdx.x * 256 + threadIdx.x;
  if (i >= n4) return;
  float4 x = *(float4*)&a[i * 4];
  float4 y = *(const float4*)&b[i * 4];
  x.x += y.x; x.y += y.y; x.z += y.z; x.w += y.w;
  *(float4*)&a[i * 4] = x;
}
__global__ __launch_bounds__(256)
void ew_mul(float* __restrict__ a, const float* __restrict__ b, size_t n4) {
  size_t i = (size_t)blockIdx.x * 256 + threadIdx.x;
  if (i >= n4) return;
  float4 x = *(float4*)&a[i * 4];
  float4 y = *(const float4*)&b[i * 4];
  x.x *= y.x; x.y *= y.y; x.z *= y.z; x.w *= y.w;
  *(float4*)&a[i * 4] = x;
}
__global__ __launch_bounds__(256)
void fillz(float* __restrict__ p, size_t n4) {
  size_t i = (size_t)blockIdx.x * 256 + threadIdx.x;
  if (i < n4) *(float4*)&p[i * 4] = float4{0.f, 0.f, 0.f, 0.f};
}

// fin (B,D,256) -> out (B,256,D)
__global__ __launch_bounds__(256)
void out_transpose(const float* __restrict__ fin, float* __restrict__ out) {
  size_t i = (size_t)blockIdx.x * 256 + threadIdx.x;
  size_t total = (size_t)BB * 256 * DD;
  if (i >= total) return;
  int d = (int)(i % DD);
  int t = (int)((i / DD) % 256);
  int b = (int)(i / ((size_t)256 * DD));
  out[i] = fin[((size_t)b * DD + d) * 256 + t];
}

// ------------------------------- host side ---------------------------------
static inline dim3 g1(size_t n) { return dim3((unsigned)((n + 255) / 256)); }

static inline void gemm(hipStream_t st, const float* A, int lda, const float* W,
                        const float* bias, float* C, int ldc,
                        size_t M, int N, int K, int fuse, __bf16* wsBt) {
  if ((M % 64 == 0) && (N % 64 == 0) && (K % 32 == 0)) {
    wt_bf16_k<<<g1((size_t)K * N), 256, 0, st>>>(W, wsBt, K, N);
    dim3 grid((unsigned)(M / 64), (unsigned)(N / 64));
    gemm_wmma_fast<<<grid, 128, 0, st>>>(A, lda, wsBt, bias, C, ldc, (int)M, N, K, fuse);
  } else {
    dim3 grid((unsigned)((M + 31) / 32), (unsigned)((N + 31) / 32));
    gemm_bf16_wmma<<<grid, 128, 0, st>>>(A, lda, W, N, bias, C, ldc, (int)M, N, K, fuse);
  }
}

struct MambaP { const float *A_log, *D, *W_dt, *W_in, *W_out, *W_x, *b_dt, *conv_b, *conv_w; };
static MambaP read_mamba(const float* const* L, int& c) {
  MambaP m;
  m.A_log = L[c++]; m.D = L[c++]; m.W_dt = L[c++]; m.W_in = L[c++]; m.W_out = L[c++];
  m.W_x = L[c++]; m.b_dt = L[c++]; m.conv_b = L[c++]; m.conv_w = L[c++];
  return m;
}

// ms layout (floats): xz[1024T] | xc[512T] | zs[512T] | proj[64T]
static void run_mamba(hipStream_t st, const float* x, float* out,
                      int Nb, int L, const MambaP& p, float* ms, __bf16* wsBt) {
  size_t T = (size_t)Nb * L;
  float* xz   = ms;
  float* xc   = ms + 1024 * T;
  float* zs   = xc + 512 * T;
  float* proj = zs + 512 * T;
  float* dt   = xz;
  float* yv   = xz + 512 * T;
  gemm(st, x, DM, p.W_in, nullptr, xz, 2 * DINNER, T, 2 * DINNER, DM, 0, wsBt);
  conv_silu_k<<<g1(T * DINNER / 4), 256, 0, st>>>(xz, p.conv_w, p.conv_b, xc, zs, Nb, L);
  gemm(st, xc, DINNER, p.W_x, nullptr, proj, 48, T, 48, DINNER, 0, wsBt);
  gemm(st, proj, 48, p.W_dt, p.b_dt, dt, DINNER, T, DINNER, DTRANK, 1 | 4, wsBt);
  mamba_scan_k<<<g1((size_t)Nb * DINNER), 256, 0, st>>>(xc, dt, proj, p.A_log, p.D, yv, Nb, L);
  ew_mul<<<g1(T * DINNER / 4), 256, 0, st>>>(yv, zs, T * DINNER / 4);
  gemm(st, yv, DINNER, p.W_out, nullptr, out, DM, T, DM, DINNER, 0, wsBt);
}

static void two_stage(hipStream_t st, float* x, float* tmp, int S,
                      const MambaP& pt, const MambaP& pd, float* ms, __bf16* wsBt) {
  size_t n4 = (size_t)BB * DD * S * (DM / 4);
  run_mamba(st, x, tmp, BB * DD, S, pt, ms, wsBt);
  permute0213<<<g1(n4), 256, 0, st>>>(tmp, x, BB, DD, S);
  run_mamba(st, x, tmp, BB * S, DD, pd, ms, wsBt);
  permute0213<<<g1(n4), 256, 0, st>>>(tmp, x, BB, S, DD);
}

extern "C" void kernel_launch(void* const* d_in, const int* in_sizes, int n_in,
                              void* d_out, int out_size, void* d_ws, size_t ws_size,
                              hipStream_t stream) {
  (void)in_sizes; (void)n_in; (void)out_size; (void)ws_size;
  const float* const* L = (const float* const*)d_in;
  int c = 0;
  const float* x_enc = L[c++];

  struct DecP {
    const float *W1, *W2, *b1, *b2, *ln1_b, *ln1_g, *ln2_b, *ln2_g, *pW, *pb;
    MambaP dim, time;
  } dec[4];
  for (int i = 0; i < 4; ++i) {
    dec[i].W1 = L[c++]; dec[i].W2 = L[c++]; dec[i].b1 = L[c++]; dec[i].b2 = L[c++];
    dec[i].dim = read_mamba(L, c);
    dec[i].ln1_b = L[c++]; dec[i].ln1_g = L[c++]; dec[i].ln2_b = L[c++]; dec[i].ln2_g = L[c++];
    dec[i].pW = L[c++]; dec[i].pb = L[c++];
    dec[i].time = read_mamba(L, c);
  }
  const float* dec_pos = L[c++];
  struct EncP { MambaP dim, time; const float *mW, *mb, *mbW, *mg; } enc[4];
  for (int i = 0; i < 4; ++i) {
    enc[i].dim = read_mamba(L, c);
    if (i > 0) { enc[i].mW = L[c++]; enc[i].mb = L[c++]; enc[i].mbW = L[c++]; enc[i].mg = L[c++]; }
    enc[i].time = read_mamba(L, c);
  }
  const float* patch_W = L[c++];
  const float* pos_emb = L[c++];

  // ---- workspace carve ----
  float* ws = (float*)d_ws;
  size_t off = 0;
  auto wsa = [&](size_t n) { float* p = ws + off; off += n; return p; };
  const size_t Tmax = (size_t)BB * DD * IN_SEG;
  __bf16* wsBt = (__bf16*)wsa(262144);
  float* ms    = wsa(2112 * Tmax);
  float* xcur  = wsa((size_t)BB * DD * IN_SEG * DM);
  float* xtmp  = wsa((size_t)BB * DD * IN_SEG * DM);
  float* enc0  = wsa((size_t)BB * DD * 16 * DM);
  float* enc1  = wsa((size_t)BB * DD * 16 * DM);
  float* enc2  = wsa((size_t)BB * DD * 8 * DM);
  float* enc3  = wsa((size_t)BB * DD * 4 * DM);
  float* xd    = wsa((size_t)BB * DD * OUT_SEG * DM);
  float* xdt   = wsa((size_t)BB * DD * OUT_SEG * DM);
  float* fin   = wsa((size_t)BB * DD * OUT_SEG * PATCH);

  // ---- patch embedding ----
  {
    size_t rows = (size_t)BB * DD * IN_SEG;
    gather_patch<<<g1(rows * PATCH), 256, 0, stream>>>(x_enc, ms);
    gemm(stream, ms, PATCH, patch_W, nullptr, xcur, DM, rows, DM, PATCH, 0, wsBt);
    rowbcast<<<g1(rows * DM / 4), 256, 0, stream>>>(xcur, pos_emb, IN_SEG, DM, rows * DM / 4, 1);
    hipMemcpyAsync(enc0, xcur, rows * DM * sizeof(float), hipMemcpyDeviceToDevice, stream);
  }

  // ---- encoder (block 3 output dead -> skipped) ----
  two_stage(stream, xcur, xtmp, 16, enc[0].time, enc[0].dim, ms, wsBt);
  hipMemcpyAsync(enc1, xcur, (size_t)BB * DD * 16 * DM * sizeof(float),
                 hipMemcpyDeviceToDevice, stream);
  {   // block 1: merge 16 -> 8
    int S2 = 8; size_t rows = (size_t)BB * DD * S2;
    seg_gather<<<g1(rows * 2 * DM / 4), 256, 0, stream>>>(xcur, xtmp, S2);
    layernorm_k<<<(unsigned)rows, 256, 0, stream>>>(xtmp, enc[1].mg, enc[1].mb, xtmp, 2 * DM);
    gemm(stream, xtmp, 2 * DM, enc[1].mW, enc[1].mbW, xcur, DM, rows, DM, 2 * DM, 1, wsBt);
    two_stage(stream, xcur, xtmp, S2, enc[1].time, enc[1].dim, ms, wsBt);
    hipMemcpyAsync(enc2, xcur, rows * DM * sizeof(float), hipMemcpyDeviceToDevice, stream);
  }
  {   // block 2: merge 8 -> 4
    int S2 = 4; size_t rows = (size_t)BB * DD * S2;
    seg_gather<<<g1(rows * 2 * DM / 4), 256, 0, stream>>>(xcur, xtmp, S2);
    layernorm_k<<<(unsigned)rows, 256, 0, stream>>>(xtmp, enc[2].mg, enc[2].mb, xtmp, 2 * DM);
    gemm(stream, xtmp, 2 * DM, enc[2].mW, enc[2].mbW, xcur, DM, rows, DM, 2 * DM, 1, wsBt);
    two_stage(stream, xcur, xtmp, S2, enc[2].time, enc[2].dim, ms, wsBt);
    hipMemcpyAsync(enc3, xcur, rows * DM * sizeof(float), hipMemcpyDeviceToDevice, stream);
  }

  // ---- decoder ----
  const size_t rows8 = (size_t)BB * DD * OUT_SEG;
  fillz<<<g1(rows8 * PATCH / 4), 256, 0, stream>>>(fin, rows8 * PATCH / 4);
  rowbcast<<<g1(rows8 * DM / 4), 256, 0, stream>>>(xd, dec_pos, OUT_SEG, DM, rows8 * DM / 4, 0);
  const float* encs[4] = {enc0, enc1, enc2, enc3};
  const int Se[4] = {16, 16, 8, 4};
  for (int i = 0; i < 4; ++i) {
    two_stage(stream, xd, xdt, OUT_SEG, dec[i].time, dec[i].dim, ms, wsBt);
    add_interp<<<g1(rows8 * DM / 4), 256, 0, stream>>>(xd, encs[i], Se[i]);
    layernorm_k<<<(unsigned)rows8, 256, 0, stream>>>(xd, dec[i].ln1_g, dec[i].ln1_b, xd, DM);
    float* y1   = ms;
    float* y2   = ms + rows8 * DM;
    float* pred = y2 + rows8 * DM;
    gemm(stream, xd, DM, dec[i].W1, dec[i].b1, y1, DM, rows8, DM, DM, 1 | 2, wsBt);
    gemm(stream, y1, DM, dec[i].W2, dec[i].b2, y2, DM, rows8, DM, DM, 1, wsBt);
    ew_add<<<g1(rows8 * DM / 4), 256, 0, stream>>>(y2, xd, rows8 * DM / 4);
    layernorm_k<<<(unsigned)rows8, 256, 0, stream>>>(y2, dec[i].ln2_g, dec[i].ln2_b, xd, DM);
    gemm(stream, xd, DM, dec[i].pW, dec[i].pb, pred, PATCH, rows8, PATCH, DM, 1, wsBt);
    ew_add<<<g1(rows8 * PATCH / 4), 256, 0, stream>>>(fin, pred, rows8 * PATCH / 4);
  }
  out_transpose<<<g1((size_t)BB * 256 * DD), 256, 0, stream>>>(fin, (float*)d_out);
}
